// SimpleLSTM_83880711291027
// MI455X (gfx1250) — compile-verified, matrix operand order checked by reading, not run on hardware
//
#include <hip/hip_runtime.h>

typedef __attribute__((ext_vector_type(16))) _Float16 v16h;
typedef __attribute__((ext_vector_type(8)))  _Float16 v8h;
typedef __attribute__((ext_vector_type(8)))  float    v8f;

constexpr int B_   = 1024;
constexpr int S_   = 256;
constexpr int H_   = 128;   // hidden
constexpr int IN_  = 2;
constexpr int OUT_ = 2;
constexpr int T_   = 512;
constexpr int BM   = 16;    // batch rows per workgroup (one WMMA M tile)
constexpr int NW   = 8;     // waves per workgroup: wave j owns hidden cols [16j,16j+16)

__device__ __forceinline__ float fast_rcp(float x) {
    return __builtin_amdgcn_rcpf(x);          // v_rcp_f32
}

#if __has_builtin(__builtin_amdgcn_tanhf)
// CDNA5 has native v_tanh_f32 (TRANS32 op)
__device__ __forceinline__ float fast_tanh(float x) {
    return __builtin_amdgcn_tanhf(x);
}
__device__ __forceinline__ float fast_sigmoid(float x) {
    // sigmoid(x) = 0.5*tanh(x/2) + 0.5   (mul + tanh + fma)
    return __builtin_fmaf(0.5f, __builtin_amdgcn_tanhf(0.5f * x), 0.5f);
}
#else
__device__ __forceinline__ float fast_sigmoid(float x) {
    // rcp(inf)=0, rcp(1)=1 -> correct saturation at extremes
    return fast_rcp(1.0f + __expf(-x));
}
__device__ __forceinline__ float fast_tanh(float x) {
    // 2*sigmoid(2x)-1; saturates to +-1 correctly
    return __builtin_fmaf(2.0f, fast_rcp(1.0f + __expf(-2.0f * x)), -1.0f);
}
#endif

__global__ __launch_bounds__(BM * NW * 2, 1)
void lstm_seq2seq_kernel(const float* __restrict__ x,
                         const float* __restrict__ w_ih,
                         const float* __restrict__ w_hh,
                         const float* __restrict__ b_ih,
                         const float* __restrict__ b_hh,
                         const float* __restrict__ w_fc,
                         const float* __restrict__ b_fc,
                         float* __restrict__ out)
{
    // Double-buffered hidden state, f16, row-major [m][k] (k = hidden index).
    __shared__ _Float16 h_lds[2][BM][H_];       // 8 KB
    __shared__ float    wfc_lds[OUT_][H_];      // 1 KB
    __shared__ float    pred_lds[BM][OUT_];     // 128 B

    const int tid  = threadIdx.x;
    const int wave = tid >> 5;          // 0..7
    const int lane = tid & 31;
    const int hi   = lane >> 4;         // 0/1 (lane half)
    const int ln   = lane & 15;         // A: row m ; B/C/D: col n
    const int b0   = blockIdx.x * BM;   // batch base of this workgroup

    // ---- init LDS ----
    for (int i = tid; i < BM * H_; i += blockDim.x)
        ((_Float16*)h_lds[0])[i] = (_Float16)0.0f;          // h0 = 0
    for (int i = tid; i < OUT_ * H_; i += blockDim.x)
        ((float*)wfc_lds)[i] = w_fc[i];
    __syncthreads();

    // ---- per-lane loop-invariant constants ----
    float biasr[4], wih0r[4], wih1r[4];
#pragma unroll
    for (int g = 0; g < 4; ++g) {
        const int ng = 16 * wave + 128 * g + ln;
        wih0r[g] = w_ih[ng * IN_ + 0];
        wih1r[g] = w_ih[ng * IN_ + 1];
        biasr[g] = b_ih[ng] + b_hh[ng];
    }
    const float bfc_r = b_fc[hi];

    // ---- register-resident B operand: w_hh.T tiles, f32 -> f16 once ----
    // Lane: n = 16*wave + 128*g + ln; halves 0..7  -> K = 32kk + 8*hi + i
    //                                halves 8..15 -> K = 32kk + 8*hi + 16 + (i-8)
    v16h Breg[4][4];
#pragma unroll
    for (int g = 0; g < 4; ++g) {
        const int ng = 16 * wave + 128 * g + ln;
        const float* wrow = w_hh + (size_t)ng * H_;
#pragma unroll
        for (int kk = 0; kk < 4; ++kk) {
            const int k0 = 32 * kk + 8 * hi;
            const float4 f0 = *(const float4*)(wrow + k0);
            const float4 f1 = *(const float4*)(wrow + k0 + 4);
            const float4 f2 = *(const float4*)(wrow + k0 + 16);
            const float4 f3 = *(const float4*)(wrow + k0 + 20);
            v16h Bt;
            Bt[0]  = (_Float16)f0.x; Bt[1]  = (_Float16)f0.y;
            Bt[2]  = (_Float16)f0.z; Bt[3]  = (_Float16)f0.w;
            Bt[4]  = (_Float16)f1.x; Bt[5]  = (_Float16)f1.y;
            Bt[6]  = (_Float16)f1.z; Bt[7]  = (_Float16)f1.w;
            Bt[8]  = (_Float16)f2.x; Bt[9]  = (_Float16)f2.y;
            Bt[10] = (_Float16)f2.z; Bt[11] = (_Float16)f2.w;
            Bt[12] = (_Float16)f3.x; Bt[13] = (_Float16)f3.y;
            Bt[14] = (_Float16)f3.z; Bt[15] = (_Float16)f3.w;
            Breg[g][kk] = Bt;
        }
    }

    // ---- cell state c: this wave owns the 16x16 chunk at hidden cols [16*wave,+16) ----
    float creg[8];
#pragma unroll
    for (int v = 0; v < 8; ++v) creg[v] = 0.0f;

    // ============================ time loop ============================
    // Step t computes h_t from h_{t-1} (in h_lds[p]).  For decoder steps
    // (t > S_), wave 0 computes pred_{t-1} = h_{t-1} @ w_fc.T CONCURRENTLY
    // with the other waves' WMMA work; the x-contribution (which needs that
    // prediction) is added to the gate accumulators after the WMMA chain.
#pragma unroll 1
    for (int t = 0; t < S_ + T_; ++t) {
        const int p = t & 1;             // read h from buffer p, write to p^1

        // (1) global input rows (encoder steps and first decoder step)
        float inx[8], iny[8];
        if (t <= S_) {
            const int tx = (t == S_) ? (S_ - 1) : t;
#pragma unroll
            for (int v = 0; v < 8; ++v) {
                const int m = v + 8 * hi;
                const float2 xv =
                    *(const float2*)(x + ((size_t)(b0 + m) * S_ + tx) * IN_);
                inx[v] = xv.x; iny[v] = xv.y;
            }
        }

        // (2) A operand: h_{t-1} (16x128 f16) from LDS in WMMA A layout.
        v16h Areg[4];
#pragma unroll
        for (int kk = 0; kk < 4; ++kk) {
            const v8h lo = *(const v8h*)&h_lds[p][ln][32 * kk + 8 * hi];
            const v8h hs = *(const v8h*)&h_lds[p][ln][32 * kk + 8 * hi + 16];
            v16h A;
#pragma unroll
            for (int i = 0; i < 8; ++i) { A[i] = lo[i]; A[8 + i] = hs[i]; }
            Areg[kk] = A;
        }

        // (3) decoder head for the PREVIOUS step, overlapped with WMMA below.
        if (t > S_ && wave == 0) {
            const int m = ln, o = hi;
            float acc = bfc_r;
#pragma unroll
            for (int k = 0; k < H_; k += 8) {
                const v8h hv = *(const v8h*)&h_lds[p][m][k];
#pragma unroll
                for (int j = 0; j < 8; ++j)
                    acc += (float)hv[j] * wfc_lds[o][k + j];
            }
            pred_lds[m][o] = acc;
            out[(size_t)(b0 + m) * T_ * OUT_ + (size_t)(t - S_ - 1) * OUT_ + o] = acc;
        }

        // (4) gates = bias + h_{t-1} @ w_hh.T  (4 tiles, K=128 chained WMMA)
        v8f gacc[4];
#pragma unroll
        for (int g = 0; g < 4; ++g) {
            v8f acc;
#pragma unroll
            for (int v = 0; v < 8; ++v) acc[v] = biasr[g];
#pragma unroll
            for (int kk = 0; kk < 4; ++kk)
                acc = __builtin_amdgcn_wmma_f32_16x16x32_f16(
                          false, Areg[kk], false, Breg[g][kk],
                          (short)0, acc, false, false);
            gacc[g] = acc;
        }

        // (5) fetch this step's input if it is the previous prediction
        if (t > S_) {
            __syncthreads();             // pred_lds published by wave 0
#pragma unroll
            for (int v = 0; v < 8; ++v) {
                const float2 pv = *(const float2*)(&pred_lds[v + 8 * hi][0]);
                inx[v] = pv.x; iny[v] = pv.y;
            }
        }

        // (6) add x @ w_ih.T, lane-local LSTM cell update, write h_t (f16)
#pragma unroll
        for (int g = 0; g < 4; ++g)
#pragma unroll
            for (int v = 0; v < 8; ++v)
                gacc[g][v] += inx[v] * wih0r[g] + iny[v] * wih1r[g];

#pragma unroll
        for (int v = 0; v < 8; ++v) {
            const float ig = fast_sigmoid(gacc[0][v]);
            const float fg = fast_sigmoid(gacc[1][v]);
            const float gg = fast_tanh(gacc[2][v]);
            const float og = fast_sigmoid(gacc[3][v]);
            const float cn = fg * creg[v] + ig * gg;
            creg[v] = cn;
            const float hn = og * fast_tanh(cn);
            h_lds[p ^ 1][v + 8 * hi][16 * wave + ln] = (_Float16)hn;
        }
        __syncthreads();                 // publish h_t
    }

    // ---- tail: prediction for the last decoder step (dt = T_-1) ----
    {
        const int pfin = (S_ + T_) & 1;  // buffer holding h_{S+T-1}
        if (wave == 0) {
            const int m = ln, o = hi;
            float acc = bfc_r;
#pragma unroll
            for (int k = 0; k < H_; k += 8) {
                const v8h hv = *(const v8h*)&h_lds[pfin][m][k];
#pragma unroll
                for (int j = 0; j < 8; ++j)
                    acc += (float)hv[j] * wfc_lds[o][k + j];
            }
            out[(size_t)(b0 + m) * T_ * OUT_ + (size_t)(T_ - 1) * OUT_ + o] = acc;
        }
    }
}

extern "C" void kernel_launch(void* const* d_in, const int* in_sizes, int n_in,
                              void* d_out, int out_size, void* d_ws, size_t ws_size,
                              hipStream_t stream) {
    const float* x    = (const float*)d_in[0];
    const float* w_ih = (const float*)d_in[1];
    const float* w_hh = (const float*)d_in[2];
    const float* b_ih = (const float*)d_in[3];
    const float* b_hh = (const float*)d_in[4];
    const float* w_fc = (const float*)d_in[5];
    const float* b_fc = (const float*)d_in[6];
    float* out = (float*)d_out;

    dim3 grid(B_ / BM);          // 64 workgroups, each owns 16 batch rows
    dim3 block(BM * NW * 2);     // 256 threads = 8 wave32
    lstm_seq2seq_kernel<<<grid, block, 0, stream>>>(x, w_ih, w_hh, b_ih, b_hh,
                                                    w_fc, b_fc, out);
}
